// Model_16217796510251
// MI455X (gfx1250) — matrix operand order repaired and checked by reading, not compiled, not run on hardware
//
#include <hip/hip_runtime.h>
#include <hip/hip_bf16.h>

// ---------------------------------------------------------------------------
// Shapes
// ---------------------------------------------------------------------------
#define BB   8
#define SS   32
#define JJ   4
#define LL   32
#define EE   500
#define HH   500
#define NII  5000
#define NTT  48
#define KP   512     // padded E/H for WMMA K
#define K2P  1024    // padded 2H
#define NPAD 5008    // padded NI rows for final GEMM B-matrix

typedef __attribute__((ext_vector_type(16))) _Float16 v16h;
typedef __attribute__((ext_vector_type(8)))  _Float16 v8h;
typedef __attribute__((ext_vector_type(8)))  float    v8f;

#define DEV static __device__ __forceinline__

DEV float sigmoidf_(float x) { return 1.0f / (1.0f + __expf(-x)); }
DEV float seluf_(float x) {
  const float a = 1.6732632423543772f, s = 1.0507009873554805f;
  return x > 0.f ? s * x : s * a * (__expf(x) - 1.0f);
}

// A fragment (16x32 f16, M x K): lane l holds row m=l&15; K = kb..kb+7 and
// 16+kb..16+kb+7 with kb = (l>>4)*8.  base points at (tile row 0, k-chunk 0).
DEV v16h load_frag_a(const _Float16* base, int ld) {
  const int l = threadIdx.x & 31;
  const _Float16* p = base + (size_t)(l & 15) * ld + ((l >> 4) << 3);
  v8h lo = *(const v8h*)(p);
  v8h hi = *(const v8h*)(p + 16);
  v16h r;
#pragma unroll
  for (int j = 0; j < 8; ++j) { r[j] = lo[j]; r[8 + j] = hi[j]; }
  return r;
}

// B fragment (32x16 f16, K x N) with B[k][n] = W[n][k] (row-major W, K contig):
// lane l holds col n=l&15; K = kb..kb+7 and 16+kb..16+kb+7 with kb = (l>>4)*8.
DEV v16h load_frag_b(const _Float16* base, int ld) {
  const int l = threadIdx.x & 31;
  const _Float16* p = base + (size_t)(l & 15) * ld + ((l >> 4) << 3);
  v8h lo = *(const v8h*)(p);
  v8h hi = *(const v8h*)(p + 16);
  v16h r;
#pragma unroll
  for (int j = 0; j < 8; ++j) { r[j] = lo[j]; r[8 + j] = hi[j]; }
  return r;
}

DEV v8f wmma16(v16h a, v16h b, v8f c) {
  return __builtin_amdgcn_wmma_f32_16x16x32_f16(false, a, false, b, (short)0, c,
                                                false, false);
}

// ---------------------------------------------------------------------------
// Conversions / gathers
// ---------------------------------------------------------------------------

// W [2000][500] f32 -> gate-padded [4*512][512] f16 (zeros in pad)
__global__ void k_conv_gate_w(const float* __restrict__ W,
                              _Float16* __restrict__ Wp) {
  size_t idx = (size_t)blockIdx.x * blockDim.x + threadIdx.x;
  if (idx >= (size_t)2048 * KP) return;
  int k  = idx & (KP - 1);
  int np = (int)(idx >> 9);
  int g  = np >> 9;
  int u  = np & 511;
  float v = (u < HH && k < EE) ? W[(size_t)(g * HH + u) * EE + k] : 0.f;
  Wp[idx] = (_Float16)v;
}

// b [2000] -> gate-padded [2048] f32
__global__ void k_conv_bias(const float* __restrict__ b, float* __restrict__ bp) {
  int idx = blockIdx.x * blockDim.x + threadIdx.x;
  if (idx >= 2048) return;
  int g = idx >> 9, u = idx & 511;
  bp[idx] = (u < HH) ? b[g * HH + u] : 0.f;
}

// lin_W [5000][1000] -> [5008][1024] f16
__global__ void k_conv_linw(const float* __restrict__ W,
                            _Float16* __restrict__ Wp) {
  size_t idx = (size_t)blockIdx.x * blockDim.x + threadIdx.x;
  if (idx >= (size_t)NPAD * K2P) return;
  int k = idx & (K2P - 1);
  int n = (int)(idx >> 10);
  float v = (n < NII && k < 2 * HH) ? W[(size_t)n * (2 * HH) + k] : 0.f;
  Wp[idx] = (_Float16)v;
}

// gather embeddings -> x sequence [nsteps][16][512] f16 (rows>=8, k>=500 zero)
// ids layout: ids[m*nsteps + step]
__global__ void k_embed(const int* __restrict__ ids,
                        const float* __restrict__ embW,
                        _Float16* __restrict__ xh, int nsteps) {
  size_t n = (size_t)nsteps * 16 * KP;
  size_t idx = (size_t)blockIdx.x * blockDim.x + threadIdx.x;
  if (idx >= n) return;
  int k = idx & (KP - 1);
  int m = (int)((idx >> 9) & 15);
  int s = (int)(idx >> 13);
  float v = 0.f;
  if (m < BB && k < EE) {
    int id = ids[m * nsteps + s];
    v = embW[(size_t)id * EE + k];
  }
  xh[idx] = (_Float16)v;
}

// ---------------------------------------------------------------------------
// Persistent LSTM (main & history): one workgroup, 8 waves, 32 hidden tiles.
// Wave w owns hidden tiles w*4..w*4+3; each tile computes 4 gate tiles (i,f,g,o)
// so gate fusion happens entirely in-registers. c-state lives in registers
// (exclusive (m,u) ownership per lane under the WMMA C layout), h lives in LDS
// (f16, double buffered).  hout[(m*nsteps+s)*500+u] for m<8,u<500.
// ---------------------------------------------------------------------------
__global__ void __launch_bounds__(256) k_lstm(const _Float16* __restrict__ xseq,
                                              const _Float16* __restrict__ Wi,
                                              const _Float16* __restrict__ Wh,
                                              const float* __restrict__ bias,
                                              float* __restrict__ hout,
                                              int nsteps) {
  __shared__ _Float16 hbuf[2][16 * KP];
  const int tid = threadIdx.x;
  for (int i = tid; i < 2 * 16 * KP; i += 256) hbuf[0][i] = (_Float16)0.f;
  __syncthreads();

  const int wave = tid >> 5;
  const int lane = tid & 31;
  const int col  = lane & 15;
  const int hi16 = lane >> 4;

  float cre[4][8];
#pragma unroll
  for (int i = 0; i < 4; ++i)
#pragma unroll
    for (int v = 0; v < 8; ++v) cre[i][v] = 0.f;

  for (int s = 0; s < nsteps; ++s) {
    const int cur = s & 1, nxt = cur ^ 1;
    const _Float16* xs = xseq + (size_t)s * 16 * KP;
    const _Float16* hb = hbuf[cur];
    _Float16* hn = hbuf[nxt];
#pragma unroll
    for (int i = 0; i < 4; ++i) {
      const int ut = wave * 4 + i;
      const int u  = ut * 16 + col;
      v8f acc[4];
#pragma unroll
      for (int g = 0; g < 4; ++g) {
        float bv = bias[g * 512 + u];
#pragma unroll
        for (int v = 0; v < 8; ++v) acc[g][v] = bv;
      }
      for (int kc = 0; kc < KP / 32; ++kc) {
        v16h a = load_frag_a(xs + kc * 32, KP);
#pragma unroll
        for (int g = 0; g < 4; ++g) {
          v16h b = load_frag_b(Wi + ((size_t)(g * 512 + ut * 16)) * KP + kc * 32, KP);
          acc[g] = wmma16(a, b, acc[g]);
        }
      }
      for (int kc = 0; kc < KP / 32; ++kc) {
        v16h a = load_frag_a(hb + kc * 32, KP);
#pragma unroll
        for (int g = 0; g < 4; ++g) {
          v16h b = load_frag_b(Wh + ((size_t)(g * 512 + ut * 16)) * KP + kc * 32, KP);
          acc[g] = wmma16(a, b, acc[g]);
        }
      }
#pragma unroll
      for (int v = 0; v < 8; ++v) {
        const int m = v + hi16 * 8;
        float zi = acc[0][v], zf = acc[1][v], zg = acc[2][v], zo = acc[3][v];
        float c = sigmoidf_(zf) * cre[i][v] + sigmoidf_(zi) * tanhf(zg);
        cre[i][v] = c;
        float h = sigmoidf_(zo) * tanhf(c);
        if (m < BB && u < HH) {
          hn[m * KP + u] = (_Float16)h;
          hout[((size_t)(m * nsteps + s)) * HH + u] = h;
        }
      }
    }
    __syncthreads();
  }
}

// ---------------------------------------------------------------------------
// Dilated LSTM: prev state gathered from full state history Hs/Cs [S][8][512].
// ---------------------------------------------------------------------------
__global__ void __launch_bounds__(256) k_lstm_dil(
    const _Float16* __restrict__ xseq, const _Float16* __restrict__ Wi,
    const _Float16* __restrict__ Wh, const float* __restrict__ bias,
    const int* __restrict__ didx, float* __restrict__ Hs,
    float* __restrict__ Cs, float* __restrict__ hout) {
  __shared__ _Float16 ph[16 * KP];
  const int tid  = threadIdx.x;
  const int wave = tid >> 5;
  const int lane = tid & 31;
  const int col  = lane & 15;
  const int hi16 = lane >> 4;

  for (int s = 0; s < SS; ++s) {
    // build previous-h (f16) from gathered history
    for (int idx = tid; idx < 16 * KP; idx += 256) {
      int m = idx >> 9, u = idx & (KP - 1);
      float v = 0.f;
      if (s > 0 && m < BB) {
        int it = didx[m * SS + s] & (SS - 1);
        v = Hs[((size_t)it * BB + m) * KP + u];
      }
      ph[idx] = (_Float16)v;
    }
    __syncthreads();

    const _Float16* xs = xseq + (size_t)s * 16 * KP;
#pragma unroll
    for (int i = 0; i < 4; ++i) {
      const int ut = wave * 4 + i;
      const int u  = ut * 16 + col;
      v8f acc[4];
#pragma unroll
      for (int g = 0; g < 4; ++g) {
        float bv = bias[g * 512 + u];
#pragma unroll
        for (int v = 0; v < 8; ++v) acc[g][v] = bv;
      }
      for (int kc = 0; kc < KP / 32; ++kc) {
        v16h a = load_frag_a(xs + kc * 32, KP);
#pragma unroll
        for (int g = 0; g < 4; ++g) {
          v16h b = load_frag_b(Wi + ((size_t)(g * 512 + ut * 16)) * KP + kc * 32, KP);
          acc[g] = wmma16(a, b, acc[g]);
        }
      }
      for (int kc = 0; kc < KP / 32; ++kc) {
        v16h a = load_frag_a(ph + kc * 32, KP);
#pragma unroll
        for (int g = 0; g < 4; ++g) {
          v16h b = load_frag_b(Wh + ((size_t)(g * 512 + ut * 16)) * KP + kc * 32, KP);
          acc[g] = wmma16(a, b, acc[g]);
        }
      }
#pragma unroll
      for (int v = 0; v < 8; ++v) {
        const int m = v + hi16 * 8;
        if (m < BB) {
          float pc = 0.f;
          if (s > 0) {
            int it = didx[m * SS + s] & (SS - 1);
            pc = Cs[((size_t)it * BB + m) * KP + u];
          }
          float zi = acc[0][v], zf = acc[1][v], zg = acc[2][v], zo = acc[3][v];
          float c = sigmoidf_(zf) * pc + sigmoidf_(zi) * tanhf(zg);
          float h = sigmoidf_(zo) * tanhf(c);
          Hs[((size_t)s * BB + m) * KP + u] = h;
          Cs[((size_t)s * BB + m) * KP + u] = c;
          if (u < HH) hout[((size_t)(m * SS + s)) * HH + u] = h;
        }
      }
    }
    __syncthreads();
  }
}

// ---------------------------------------------------------------------------
// pooled[b][u] = sum_s out*mask / sum_s mask
// ---------------------------------------------------------------------------
__global__ void k_pool(const float* __restrict__ out,
                       const float* __restrict__ mask,
                       float* __restrict__ pooled) {
  int idx = blockIdx.x * blockDim.x + threadIdx.x;
  if (idx >= BB * HH) return;
  int b = idx / HH, u = idx - b * HH;
  float num = 0.f, den = 0.f;
  for (int s = 0; s < SS; ++s) {
    float mk = mask[b * SS + s];
    num += out[((size_t)(b * SS + s)) * HH + u] * mk;
    den += mk;
  }
  pooled[idx] = num / den;
}

// ---------------------------------------------------------------------------
// Per (b,j): w = softmax(tim_sim[timid][tms]); sess = w @ he; avgd = mean dist
// ---------------------------------------------------------------------------
__global__ void __launch_bounds__(256) k_sess(
    const float* __restrict__ he, const float* __restrict__ tim_sim,
    const float* __restrict__ poi_dist, const int* __restrict__ seq_tim,
    const int* __restrict__ item_vec, const int* __restrict__ hist_tims,
    const int* __restrict__ hist_items, float* __restrict__ sess,
    float* __restrict__ avgd) {
  const int bj = blockIdx.x;
  const int b  = bj >> 2;
  __shared__ float w[SS - 1][LL];
  __shared__ int tms[LL], hits[LL];
  if (threadIdx.x < LL) {
    tms[threadIdx.x]  = hist_tims[bj * LL + threadIdx.x];
    hits[threadIdx.x] = hist_items[bj * LL + threadIdx.x];
  }
  __syncthreads();
  for (int s = threadIdx.x; s < SS - 1; s += 256) {
    int timid = seq_tim[b * SS + s];
    const float* row = tim_sim + (size_t)timid * NTT;
    float vals[LL];
    float mx = -1e30f;
    for (int l = 0; l < LL; ++l) { vals[l] = row[tms[l]]; mx = fmaxf(mx, vals[l]); }
    float sum = 0.f;
    for (int l = 0; l < LL; ++l) { vals[l] = __expf(vals[l] - mx); sum += vals[l]; }
    float inv = 1.f / sum;
    for (int l = 0; l < LL; ++l) w[s][l] = vals[l] * inv;
    int poiid = item_vec[b * SS + s];
    const float* drow = poi_dist + (size_t)poiid * NII;
    float a = 0.f;
    for (int l = 0; l < LL; ++l) a += drow[hits[l]];
    avgd[bj * (SS - 1) + s] = a * (1.f / (float)LL);
  }
  __syncthreads();
  const float* heb = he + (size_t)bj * LL * HH;
  float* sb = sess + (size_t)bj * (SS - 1) * HH;
  for (int idx = threadIdx.x; idx < (SS - 1) * HH; idx += 256) {
    int s = idx / HH, u = idx - s * HH;
    float acc = 0.f;
    for (int l = 0; l < LL; ++l) acc += w[s][l] * heb[(size_t)l * HH + u];
    sb[idx] = acc;
  }
}

// ---------------------------------------------------------------------------
// Per (b,s<31): sims->attn->lin1+selu->layer2->l2s->ol2->selu -> y[b][s][:]
// ---------------------------------------------------------------------------
__global__ void __launch_bounds__(256) k_attn(
    const float* __restrict__ sess, const float* __restrict__ pooled,
    const float* __restrict__ avgd, const float* __restrict__ out_hs,
    const float* __restrict__ lin1W, const float* __restrict__ lin1b,
    float* __restrict__ y) {
  const int bs = blockIdx.x;
  const int b = bs / (SS - 1), s = bs % (SS - 1);
  const int tid = threadIdx.x;
  __shared__ float red[256];
  __shared__ float attn[HH];
  __shared__ float l2[HH];
  const float* sb = sess + ((size_t)b * JJ * (SS - 1) + s) * HH;
  const float* pb = pooled + b * HH;

  float dots[4];
#pragma unroll
  for (int j = 0; j < 4; ++j) {
    float pp = 0.f;
    for (int u = tid; u < HH; u += 256)
      pp += sb[(size_t)j * (SS - 1) * HH + u] * pb[u];
    red[tid] = pp; __syncthreads();
    for (int off = 128; off > 0; off >>= 1) {
      if (tid < off) red[tid] += red[tid + off];
      __syncthreads();
    }
    dots[j] = red[0];
    __syncthreads();
  }
  float mx = fmaxf(fmaxf(dots[0], dots[1]), fmaxf(dots[2], dots[3]));
  float e0 = __expf(dots[0] - mx), e1 = __expf(dots[1] - mx);
  float e2 = __expf(dots[2] - mx), e3 = __expf(dots[3] - mx);
  float inv = 1.f / (e0 + e1 + e2 + e3);
  float sims[4] = {e0 * inv, e1 * inv, e2 * inv, e3 * inv};
  for (int u = tid; u < HH; u += 256) {
    float a = 0.f;
#pragma unroll
    for (int j = 0; j < 4; ++j) a += sims[j] * sb[(size_t)j * (SS - 1) * HH + u];
    attn[u] = a;
  }
  __syncthreads();
  for (int u = tid; u < HH; u += 256) {
    float acc = lin1b[u];
    const float* wr = lin1W + (size_t)u * HH;
    for (int k = 0; k < HH; ++k) acc += attn[k] * wr[k];
    l2[u] = 0.5f * seluf_(acc) + 0.5f * out_hs[((size_t)(b * SS + s)) * HH + u];
  }
  __syncthreads();
#pragma unroll
  for (int j = 0; j < 4; ++j) {
    float pp = 0.f;
    for (int u = tid; u < HH; u += 256)
      pp += sb[(size_t)j * (SS - 1) * HH + u] * l2[u];
    red[tid] = pp; __syncthreads();
    for (int off = 128; off > 0; off >>= 1) {
      if (tid < off) red[tid] += red[tid + off];
      __syncthreads();
    }
    dots[j] = red[0] / avgd[(b * JJ + j) * (SS - 1) + s];
    __syncthreads();
  }
  mx = fmaxf(fmaxf(dots[0], dots[1]), fmaxf(dots[2], dots[3]));
  e0 = __expf(dots[0] - mx); e1 = __expf(dots[1] - mx);
  e2 = __expf(dots[2] - mx); e3 = __expf(dots[3] - mx);
  inv = 1.f / (e0 + e1 + e2 + e3);
  float l2s[4] = {e0 * inv, e1 * inv, e2 * inv, e3 * inv};
  for (int u = tid; u < HH; u += 256) {
    float a = 0.f;
#pragma unroll
    for (int j = 0; j < 4; ++j) a += l2s[j] * sb[(size_t)j * (SS - 1) * HH + u];
    y[((size_t)(b * SS + s)) * HH + u] = seluf_(a);
  }
}

// ---------------------------------------------------------------------------
// feat [256][1024] f16: [y | 0.5*(selu(out)+selu(out_hie)) | zero-pad]
// ---------------------------------------------------------------------------
__global__ void k_feat(const float* __restrict__ y,
                       const float* __restrict__ out_hs,
                       const float* __restrict__ out_hie,
                       _Float16* __restrict__ feat) {
  size_t idx = (size_t)blockIdx.x * blockDim.x + threadIdx.x;
  if (idx >= (size_t)BB * SS * K2P) return;
  int cpos = idx & (K2P - 1);
  int r = (int)(idx >> 10);
  int s = r & (SS - 1);
  float v = 0.f;
  if (cpos < HH) {
    if (s < SS - 1) v = y[(size_t)r * HH + cpos];
  } else if (cpos < 2 * HH) {
    size_t o = (size_t)r * HH + (cpos - HH);
    v = 0.5f * (seluf_(out_hs[o]) + seluf_(out_hie[o]));
  }
  feat[idx] = (_Float16)v;
}

// ---------------------------------------------------------------------------
// logits = feat @ lin_W.T + lin_b   (WMMA, M=256, K=1024, N=5008->5000)
// grid (16, 40), block 256 = 8 waves, one N-tile per wave.
// ---------------------------------------------------------------------------
__global__ void __launch_bounds__(256) k_final(const _Float16* __restrict__ feat,
                                               const _Float16* __restrict__ Wl,
                                               const float* __restrict__ lb,
                                               float* __restrict__ logits) {
  const int mt = blockIdx.x;
  const int nt = blockIdx.y * 8 + (threadIdx.x >> 5);
  if (nt >= NPAD / 16) return;
  const int lane = threadIdx.x & 31;
  const int col = lane & 15, hi16 = lane >> 4;
  const int n = nt * 16 + col;
  v8f acc;
  float bv = (n < NII) ? lb[n] : 0.f;
#pragma unroll
  for (int v = 0; v < 8; ++v) acc[v] = bv;
  const _Float16* fa = feat + (size_t)mt * 16 * K2P;
  const _Float16* wb = Wl + (size_t)nt * 16 * K2P;
  for (int kc = 0; kc < K2P / 32; ++kc) {
    v16h a = load_frag_a(fa + kc * 32, K2P);
    v16h b = load_frag_b(wb + kc * 32, K2P);
    acc = wmma16(a, b, acc);
  }
  if (n < NII) {
#pragma unroll
    for (int v = 0; v < 8; ++v) {
      int m = mt * 16 + v + hi16 * 8;
      logits[(size_t)m * NII + n] = acc[v];
    }
  }
}

// ---------------------------------------------------------------------------
// in-place log_softmax over rows of 5000
// ---------------------------------------------------------------------------
__global__ void __launch_bounds__(256) k_lsm(float* __restrict__ logits) {
  __shared__ float red[256];
  const int tid = threadIdx.x;
  float* row = logits + (size_t)blockIdx.x * NII;
  float mx = -1e30f;
  for (int i = tid; i < NII; i += 256) mx = fmaxf(mx, row[i]);
  red[tid] = mx; __syncthreads();
  for (int off = 128; off > 0; off >>= 1) {
    if (tid < off) red[tid] = fmaxf(red[tid], red[tid + off]);
    __syncthreads();
  }
  mx = red[0]; __syncthreads();
  float sm = 0.f;
  for (int i = tid; i < NII; i += 256) sm += __expf(row[i] - mx);
  red[tid] = sm; __syncthreads();
  for (int off = 128; off > 0; off >>= 1) {
    if (tid < off) red[tid] += red[tid + off];
    __syncthreads();
  }
  sm = red[0]; __syncthreads();
  float lse = mx + __logf(sm);
  for (int i = tid; i < NII; i += 256) row[i] = row[i] - lse;
}

// ---------------------------------------------------------------------------
// Host launcher
// ---------------------------------------------------------------------------
extern "C" void kernel_launch(void* const* d_in, const int* in_sizes, int n_in,
                              void* d_out, int out_size, void* d_ws,
                              size_t ws_size, hipStream_t stream) {
  (void)in_sizes; (void)n_in; (void)out_size; (void)ws_size;
  const int*   item_vectors = (const int*)d_in[0];
  const int*   sequence_tim = (const int*)d_in[1];
  const int*   dilated_idx  = (const int*)d_in[2];
  const int*   hist_items   = (const int*)d_in[3];
  const int*   hist_tims    = (const int*)d_in[4];
  const float* mask         = (const float*)d_in[5];
  const float* tim_sim      = (const float*)d_in[6];
  const float* poi_dist     = (const float*)d_in[7];
  const float* emb_W        = (const float*)d_in[8];
  const float* lstm_Wi      = (const float*)d_in[9];
  const float* lstm_Wh      = (const float*)d_in[10];
  const float* lstm_b       = (const float*)d_in[11];
  const float* hist_Wi      = (const float*)d_in[12];
  const float* hist_Wh      = (const float*)d_in[13];
  const float* hist_b       = (const float*)d_in[14];
  const float* dil_Wi       = (const float*)d_in[15];
  const float* dil_Wh       = (const float*)d_in[16];
  const float* dil_b        = (const float*)d_in[17];
  const float* lin1_W       = (const float*)d_in[18];
  const float* lin1_b       = (const float*)d_in[19];
  const float* lin_W        = (const float*)d_in[20];
  const float* lin_b        = (const float*)d_in[21];
  float* logits = (float*)d_out;

  char* p = (char*)d_ws;
  auto alloc = [&](size_t bytes) {
    void* r = (void*)p;
    p += (bytes + 255) & ~(size_t)255;
    return r;
  };
  _Float16* items_h = (_Float16*)alloc((size_t)SS * 16 * KP * 2);
  _Float16* histx_h = (_Float16*)alloc((size_t)JJ * LL * 16 * KP * 2);
  _Float16* WiM = (_Float16*)alloc((size_t)2048 * KP * 2);
  _Float16* WhM = (_Float16*)alloc((size_t)2048 * KP * 2);
  _Float16* WiH = (_Float16*)alloc((size_t)2048 * KP * 2);
  _Float16* WhH = (_Float16*)alloc((size_t)2048 * KP * 2);
  _Float16* WiD = (_Float16*)alloc((size_t)2048 * KP * 2);
  _Float16* WhD = (_Float16*)alloc((size_t)2048 * KP * 2);
  float* bM = (float*)alloc(2048 * 4);
  float* bH = (float*)alloc(2048 * 4);
  float* bD = (float*)alloc(2048 * 4);
  float* out_hs  = (float*)alloc((size_t)BB * SS * HH * 4);
  float* out_hie = (float*)alloc((size_t)BB * SS * HH * 4);
  float* Hs = (float*)alloc((size_t)SS * BB * KP * 4);
  float* Cs = (float*)alloc((size_t)SS * BB * KP * 4);
  float* he = (float*)alloc((size_t)BB * JJ * LL * HH * 4);
  float* sess = (float*)alloc((size_t)BB * JJ * (SS - 1) * HH * 4);
  float* avgd = (float*)alloc((size_t)BB * JJ * (SS - 1) * 4);
  float* pooled = (float*)alloc((size_t)BB * HH * 4);
  float* ybuf = (float*)alloc((size_t)BB * SS * HH * 4);
  _Float16* feat = (_Float16*)alloc((size_t)BB * SS * K2P * 2);
  _Float16* WlP  = (_Float16*)alloc((size_t)NPAD * K2P * 2);

  // weight / bias conversions (gate-padded f16 layouts)
  k_conv_gate_w<<<4096, 256, 0, stream>>>(lstm_Wi, WiM);
  k_conv_gate_w<<<4096, 256, 0, stream>>>(lstm_Wh, WhM);
  k_conv_gate_w<<<4096, 256, 0, stream>>>(hist_Wi, WiH);
  k_conv_gate_w<<<4096, 256, 0, stream>>>(hist_Wh, WhH);
  k_conv_gate_w<<<4096, 256, 0, stream>>>(dil_Wi, WiD);
  k_conv_gate_w<<<4096, 256, 0, stream>>>(dil_Wh, WhD);
  k_conv_bias<<<8, 256, 0, stream>>>(lstm_b, bM);
  k_conv_bias<<<8, 256, 0, stream>>>(hist_b, bH);
  k_conv_bias<<<8, 256, 0, stream>>>(dil_b, bD);
  k_conv_linw<<<(int)(((size_t)NPAD * K2P + 255) / 256), 256, 0, stream>>>(lin_W, WlP);

  // embeddings
  k_embed<<<(SS * 16 * KP + 255) / 256, 256, 0, stream>>>(item_vectors, emb_W,
                                                          items_h, SS);
  k_embed<<<(JJ * LL * 16 * KP + 255) / 256, 256, 0, stream>>>(hist_items, emb_W,
                                                               histx_h, JJ * LL);

  // three LSTMs (persistent single-workgroup WMMA kernels)
  k_lstm<<<1, 256, 0, stream>>>(items_h, WiM, WhM, bM, out_hs, SS);
  k_lstm<<<1, 256, 0, stream>>>(histx_h, WiH, WhH, bH, he, JJ * LL);
  k_lstm_dil<<<1, 256, 0, stream>>>(items_h, WiD, WhD, bD, dilated_idx, Hs, Cs,
                                    out_hie);

  // pooled session representation
  k_pool<<<(BB * HH + 255) / 256, 256, 0, stream>>>(out_hs, mask, pooled);

  // per-(b,j) session attention over history LSTM outputs
  k_sess<<<BB * JJ, 256, 0, stream>>>(he, tim_sim, poi_dist, sequence_tim,
                                      item_vectors, hist_tims, hist_items, sess,
                                      avgd);

  // per-(b,s) two-level attention + lin1 + selu
  k_attn<<<BB * (SS - 1), 256, 0, stream>>>(sess, pooled, avgd, out_hs, lin1_W,
                                            lin1_b, ybuf);

  // features, final projection (WMMA), log-softmax
  k_feat<<<(BB * SS * K2P + 255) / 256, 256, 0, stream>>>(ybuf, out_hs, out_hie,
                                                          feat);
  k_final<<<dim3(BB * SS / 16, (NPAD / 16 + 7) / 8), 256, 0, stream>>>(feat, WlP,
                                                                       lin_b,
                                                                       logits);
  k_lsm<<<BB * SS, 256, 0, stream>>>(logits);
}
